// SFT_23527830848169
// MI455X (gfx1250) — compile-verified, matrix operand order checked by reading, not run on hardware
//
#include <hip/hip_runtime.h>
#include <hip/hip_bf16.h>

typedef __attribute__((ext_vector_type(16))) __bf16 v16bf;
typedef __attribute__((ext_vector_type(8)))  float  v8f;
typedef unsigned int v4u  __attribute__((ext_vector_type(4)));
typedef int          v8i_t __attribute__((ext_vector_type(8)));
typedef int          v4i_t __attribute__((ext_vector_type(4)));

#define BN_EPS 1e-5f

// ---- feature detection (probe-verified facts from CDNA5_HIP.md) -----------
#if defined(__has_builtin)
#  if __has_builtin(__builtin_amdgcn_tensor_load_to_lds)
#    define HAVE_TDM 1
#  endif
#  if __has_builtin(__builtin_amdgcn_s_wait_tensorcnt)
#    define HAVE_TCNT 1
#  endif
#endif
#if defined(__has_include)
#  if __has_include(<hip/amd_detail/amd_gfx1250_TDM.h>)
#    define TDM_ARITY6 1   // amdgpu-toolchain (clang-23): 6-arg builtin
#  endif
#endif

// Compile-time probes: these show up in compiler stderr so the next round can
// tell definitively which data-movement path was selected.
#if defined(HAVE_TDM)
#  warning "CDNA5 probe: TDM tensor_load_to_lds path ENABLED"
#else
#  warning "CDNA5 probe: TDM not available - manual LDS copy fallback"
#endif
#if defined(TDM_ARITY6)
#  warning "CDNA5 probe: TDM builtin arity-6 (amdgpu-toolchain)"
#endif

__device__ __forceinline__ unsigned short f2bf(float f) {
    unsigned int u = __float_as_uint(f);
    u += 0x7fffu + ((u >> 16) & 1u);           // round-to-nearest-even
    return (unsigned short)(u >> 16);
}
__device__ __forceinline__ float bf2f(unsigned short s) {
    return __uint_as_float(((unsigned int)s) << 16);
}

union FragU { v16bf v; uint4 q[2]; };

// 16x32 bf16 WMMA operand fragment from a row-major LDS tile.
// Lane l: row = l&15, K-chunks {base, base+16}, base = (l&16)?8:0.
__device__ __forceinline__ v16bf ld_frag(const unsigned short* base, int pitch,
                                         int row0, int k0, int lane) {
    int r  = lane & 15;
    int kb = (lane & 16) ? 8 : 0;
    const unsigned short* p = base + (size_t)(row0 + r) * pitch + k0 + kb;
    FragU f;
    f.q[0] = *(const uint4*)(p);
    f.q[1] = *(const uint4*)(p + 16);
    return f.v;
}

__device__ __forceinline__ v8f wmma_bf16(v16bf a, v16bf b, v8f c) {
    return __builtin_amdgcn_wmma_f32_16x16x32_bf16(false, a, false, b,
                                                   (short)0, c, false, false);
}
__device__ __forceinline__ v8f v8f_zero() {
    v8f z = {0.f,0.f,0.f,0.f,0.f,0.f,0.f,0.f};
    return z;
}

#if defined(HAVE_TDM)
__device__ __forceinline__ unsigned lds_addr_of(const void* p) {
    return (unsigned)(unsigned long long)
           (__attribute__((address_space(3))) const void*)p;
}
// 2D TDM load: tile (tile_d0 x tile_d1) of 2-byte elements from a row-major
// tensor (row length tensor_d0, row stride stride0, tensor_d1 rows) into LDS,
// with optional per-interval DWORD padding (gives padded LDS pitch).
__device__ __forceinline__ void tdm_load_2d(unsigned lds_off, const void* gptr,
                                            unsigned tensor_d0, unsigned tensor_d1,
                                            unsigned tile_d0, unsigned tile_d1,
                                            unsigned stride0,
                                            unsigned pad_en, unsigned pad_itv,
                                            unsigned pad_amt) {
    unsigned long long ga = (unsigned long long)gptr;
    v4u g0;
    g0[0] = 1u;                                        // count=1, user mode
    g0[1] = lds_off;                                   // lds_addr
    g0[2] = (unsigned)ga;                              // global_addr[31:0]
    g0[3] = (unsigned)((ga >> 32) & 0x1ffffffu)        // global_addr[56:32]
          | 0x80000000u;                               // type=2 ("image")
    v8i_t g1;
    g1[0] = (int)((1u << 16) |                         // data_size = 2 bytes
                  (pad_en ? ((1u << 20) | (pad_itv << 22) | (pad_amt << 25))
                          : 0u));
    g1[1] = (int)((tensor_d0 & 0xffffu) << 16);                        // dim0 lo
    g1[2] = (int)(((tensor_d0 >> 16) & 0xffffu) |
                  ((tensor_d1 & 0xffffu) << 16));                      // dim0 hi|dim1 lo
    g1[3] = (int)(((tensor_d1 >> 16) & 0xffffu) |
                  ((tile_d0 & 0xffffu) << 16));                        // dim1 hi|tile0
    g1[4] = (int)(tile_d1 & 0xffffu);                                  // tile1, tile2=0
    g1[5] = (int)stride0;                                              // stride0 lo
    g1[6] = 0;
    g1[7] = 0;
    v4i_t z4 = {0, 0, 0, 0};
#if defined(TDM_ARITY6)
    v8i_t z8 = {0, 0, 0, 0, 0, 0, 0, 0};
    __builtin_amdgcn_tensor_load_to_lds(g0, g1, z4, z4, z8, 0);
#else
    __builtin_amdgcn_tensor_load_to_lds(g0, g1, z4, z4, 0);
#endif
}
__device__ __forceinline__ void tdm_wait0() {
#if defined(HAVE_TCNT)
    __builtin_amdgcn_s_wait_tensorcnt((short)0);
#else
    asm volatile("s_wait_tensorcnt 0x0" ::: "memory");
#endif
}
#endif // HAVE_TDM

// ---------------------------------------------------------------- utilities
__global__ void zero_f32(float* p, int n) {
    int i = blockIdx.x * blockDim.x + threadIdx.x;
    if (i < n) p[i] = 0.f;
}

__global__ void f32_to_bf16(const float* __restrict__ src,
                            unsigned short* __restrict__ dst, size_t n) {
    size_t i = (size_t)blockIdx.x * blockDim.x + threadIdx.x;
    size_t stride = (size_t)gridDim.x * blockDim.x;
    for (; i < n; i += stride) dst[i] = f2bf(src[i]);
}

// ------------------------------------------------- GEMM + BN-stats epilogue
// C[M,N] = A[M,K] (bf16 rm) * Bw[N,K]^T (bf16 rm). Block tile 128x128,
// K-step 32, double-buffered LDS. Staging: TDM (wave 0 issues 2 tensor DMAs
// for tile t+1 while all waves compute tile t, then drains TENSORcnt) or, as
// fallback, per-thread global->VGPR prefetch + ds_store. 8 waves, 2x4 tiles.
template<int OUT_F32>
__global__ __launch_bounds__(256)
void gemm_bn(const unsigned short* __restrict__ A,
             const unsigned short* __restrict__ Bw,
             unsigned short* __restrict__ outB, float* __restrict__ outF,
             float* __restrict__ csum, float* __restrict__ csq,
             int K, int N) {
    __shared__ unsigned short sA[2][128 * 40];
    __shared__ unsigned short sB[2][128 * 40];

    const int tid  = threadIdx.x;
    const int lane = tid & 31;
    const int wv   = tid >> 5;
    const int wr   = wv >> 1;
    const int wc   = wv & 1;
    const int gx   = blockIdx.x;
    const int gy   = blockIdx.y;
    const int nt   = K >> 5;

    v8f acc[2][4];
#pragma unroll
    for (int i = 0; i < 2; ++i)
#pragma unroll
        for (int j = 0; j < 4; ++j) acc[i][j] = v8f_zero();

#if defined(HAVE_TDM)
    const unsigned short* ga = A  + (size_t)(gy * 128) * K;
    const unsigned short* gb = Bw + (size_t)(gx * 128) * K;
    if (wv == 0) {   // stage tile 0: 128x32 tiles, padded to pitch-40 rows
        tdm_load_2d(lds_addr_of(sA[0]), ga, 32, 128, 32, 128, (unsigned)K, 1, 3, 3);
        tdm_load_2d(lds_addr_of(sB[0]), gb, 32, 128, 32, 128, (unsigned)K, 1, 3, 3);
        tdm_wait0();
    }
    __syncthreads();
#else
    const int r0 = tid >> 2;                 // 0..63
    const int c0 = (tid & 3) << 3;           // 0,8,16,24
    const unsigned short* pa = A  + (size_t)(gy * 128 + r0) * K + c0;
    const unsigned short* pb = Bw + (size_t)(gx * 128 + r0) * K + c0;
    const size_t rowoff = (size_t)64 * K;
    uint4 ra0 = *(const uint4*)(pa);
    uint4 ra1 = *(const uint4*)(pa + rowoff);
    uint4 rb0 = *(const uint4*)(pb);
    uint4 rb1 = *(const uint4*)(pb + rowoff);
    *(uint4*)&sA[0][r0 * 40 + c0]        = ra0;
    *(uint4*)&sA[0][(r0 + 64) * 40 + c0] = ra1;
    *(uint4*)&sB[0][r0 * 40 + c0]        = rb0;
    *(uint4*)&sB[0][(r0 + 64) * 40 + c0] = rb1;
    __syncthreads();
#endif

    for (int t = 0; t < nt; ++t) {
        const int cur = t & 1, nxt = cur ^ 1;
#if defined(HAVE_TDM)
        if ((t + 1 < nt) && wv == 0) {   // async DMA of next K-tile
            tdm_load_2d(lds_addr_of(sA[nxt]), ga + (size_t)(t + 1) * 32,
                        32, 128, 32, 128, (unsigned)K, 1, 3, 3);
            tdm_load_2d(lds_addr_of(sB[nxt]), gb + (size_t)(t + 1) * 32,
                        32, 128, 32, 128, (unsigned)K, 1, 3, 3);
        }
#else
        if (t + 1 < nt) {
            const unsigned short* qa = pa + (size_t)(t + 1) * 32;
            const unsigned short* qb = pb + (size_t)(t + 1) * 32;
            ra0 = *(const uint4*)(qa);
            ra1 = *(const uint4*)(qa + rowoff);
            rb0 = *(const uint4*)(qb);
            rb1 = *(const uint4*)(qb + rowoff);
        }
#endif
        v16bf af[2], bfr[4];
#pragma unroll
        for (int i = 0; i < 2; ++i)
            af[i]  = ld_frag(sA[cur], 40, wr * 32 + i * 16, 0, lane);
#pragma unroll
        for (int j = 0; j < 4; ++j)
            bfr[j] = ld_frag(sB[cur], 40, wc * 64 + j * 16, 0, lane);
#pragma unroll
        for (int i = 0; i < 2; ++i)
#pragma unroll
            for (int j = 0; j < 4; ++j)
                acc[i][j] = wmma_bf16(af[i], bfr[j], acc[i][j]);
#if defined(HAVE_TDM)
        if ((t + 1 < nt) && wv == 0) tdm_wait0();
#else
        if (t + 1 < nt) {
            *(uint4*)&sA[nxt][r0 * 40 + c0]        = ra0;
            *(uint4*)&sA[nxt][(r0 + 64) * 40 + c0] = ra1;
            *(uint4*)&sB[nxt][r0 * 40 + c0]        = rb0;
            *(uint4*)&sB[nxt][(r0 + 64) * 40 + c0] = rb1;
        }
#endif
        __syncthreads();
    }

    const int ln = lane & 15;
    const int lh = (lane & 16) ? 8 : 0;
#pragma unroll
    for (int i = 0; i < 2; ++i) {
#pragma unroll
        for (int j = 0; j < 4; ++j) {
            int f = gx * 128 + wc * 64 + j * 16 + ln;
            float cs = 0.f, cq = 0.f;
#pragma unroll
            for (int r = 0; r < 8; ++r) {
                float v = acc[i][j][r];
                size_t m = (size_t)gy * 128 + wr * 32 + i * 16 + lh + r;
                if (OUT_F32) outF[m * N + f] = v;
                else         outB[m * N + f] = f2bf(v);
                cs += v; cq += v * v;
            }
            atomicAdd(&csum[f], cs);
            atomicAdd(&csq[f],  cq);
        }
    }
}

// ------------------------------------------------ fold BN stats into affine
__global__ void finalize_bn(const float* __restrict__ csum,
                            const float* __restrict__ csq,
                            const float* __restrict__ g,
                            const float* __restrict__ b,
                            float2* __restrict__ prm, int n, float invM) {
    int i = blockIdx.x * blockDim.x + threadIdx.x;
    if (i < n) {
        float mu  = csum[i] * invM;
        float var = csq[i] * invM - mu * mu;
        float s   = g[i] * rsqrtf(var + BN_EPS);
        prm[i]    = make_float2(s, b[i] - mu * s);
    }
}

// ------------------- apply BN affine + repack q/k/v into attention layouts
// qb : [B*H][256][64]   kb : [B*H][208][64]   vtb: [B*H][128][224] (V^T)
__global__ void repack_qkv(const unsigned short* __restrict__ ykv,
                           const unsigned short* __restrict__ yq,
                           const float2* __restrict__ prm,
                           unsigned short* __restrict__ qb,
                           unsigned short* __restrict__ kb,
                           unsigned short* __restrict__ vtb) {
    size_t i0 = (size_t)blockIdx.x * blockDim.x + threadIdx.x;
    size_t stride = (size_t)gridDim.x * blockDim.x;

    for (size_t i = i0; i < (size_t)1024 * 256 * 64; i += stride) {
        int j = (int)(i & 63); int n = (int)((i >> 6) & 255); int bh = (int)(i >> 14);
        unsigned short o = 0;
        if (n < 196) {
            int f = (bh & 15) * 64 + j;
            float2 p = prm[3072 + f];
            float v = bf2f(yq[((size_t)(bh >> 4) * 196 + n) * 1024 + f]);
            o = f2bf(p.x * v + p.y);
        }
        qb[i] = o;
    }
    for (size_t i = i0; i < (size_t)1024 * 208 * 64; i += stride) {
        int j = (int)(i & 63); size_t t = i >> 6;
        int n = (int)(t % 208); int bh = (int)(t / 208);
        unsigned short o = 0;
        if (n < 196) {
            int f = (bh & 15) * 192 + j;
            float2 p = prm[f];
            float v = bf2f(ykv[((size_t)(bh >> 4) * 196 + n) * 3072 + f]);
            o = f2bf(p.x * v + p.y);
        }
        kb[i] = o;
    }
    for (size_t i = i0; i < (size_t)1024 * 128 * 224; i += stride) {
        int n = (int)(i % 224); size_t t = i / 224;
        int dj = (int)(t & 127); int bh = (int)(t >> 7);
        unsigned short o = 0;
        if (n < 196) {
            int f = (bh & 15) * 192 + 64 + dj;
            float2 p = prm[f];
            float v = bf2f(ykv[((size_t)(bh >> 4) * 196 + n) * 3072 + f]);
            o = f2bf(p.x * v + p.y);
        }
        vtb[i] = o;
    }
}

// -------------------------------------------------------- fused attention
// grid = (4 q-tiles of 64 rows, B*H). 128 threads = 4 waves, one 16-row
// Q tile per wave. Q/K/V^T staged in LDS via TDM when available.
__global__ __launch_bounds__(128)
void attn_kernel(const unsigned short* __restrict__ qb,
                 const unsigned short* __restrict__ kb,
                 const unsigned short* __restrict__ vtb,
                 const float* __restrict__ bias,
                 unsigned short* __restrict__ hs) {
    extern __shared__ char smem[];
    unsigned short* sQ = (unsigned short*)smem;                     // 64x72
    unsigned short* sK = (unsigned short*)(smem + 9216);            // 208x72
    unsigned short* sV = (unsigned short*)(smem + 9216 + 29952);    // 128x224
    float*          sS = (float*)(smem + 9216 + 29952 + 57344);     // 4x16x224

    const int tid  = threadIdx.x;
    const int lane = tid & 31;
    const int w    = tid >> 5;
    const int bh   = blockIdx.y;
    const int b    = bh >> 4, h = bh & 15;
    const int qt   = blockIdx.x;

    const unsigned short* qg = qb  + ((size_t)bh * 256 + qt * 64) * 64;
    const unsigned short* kg = kb  + (size_t)bh * 208 * 64;
    const unsigned short* vg = vtb + (size_t)bh * 128 * 224;

#if defined(HAVE_TDM)
    if (w == 0) {   // one wave issues the three tensor DMAs, waits TENSORcnt
        tdm_load_2d(lds_addr_of(sQ), qg,  64,  64,  64,  64,  64, 1, 4, 3);
        tdm_load_2d(lds_addr_of(sK), kg,  64, 208,  64, 208,  64, 1, 4, 3);
        tdm_load_2d(lds_addr_of(sV), vg, 224, 128, 224, 128, 224, 0, 0, 0);
        tdm_wait0();
    }
#else
    for (int c = tid; c < 64 * 8; c += 128) {
        int row = c >> 3, c8 = (c & 7) << 3;
        *(uint4*)&sQ[row * 72 + c8] = *(const uint4*)&qg[row * 64 + c8];
    }
    for (int c = tid; c < 208 * 8; c += 128) {
        int row = c >> 3, c8 = (c & 7) << 3;
        *(uint4*)&sK[row * 72 + c8] = *(const uint4*)&kg[row * 64 + c8];
    }
    for (int c = tid; c < 128 * 28; c += 128) {
        *(uint4*)&sV[c * 8] = *(const uint4*)&vg[c * 8];   // same pitch (224)
    }
#endif
    __syncthreads();

    // ---- S = q k^T (16x208 per wave); batched B-frag loads
    v8f sacc[13];
#pragma unroll
    for (int j = 0; j < 13; ++j) sacc[j] = v8f_zero();
#pragma unroll
    for (int kt = 0; kt < 2; ++kt) {
        v16bf a = ld_frag(sQ, 72, w * 16, kt * 32, lane);
#pragma unroll
        for (int g = 0; g < 3; ++g) {
            v16bf bb[4];
#pragma unroll
            for (int u = 0; u < 4; ++u)
                bb[u] = ld_frag(sK, 72, (g * 4 + u) * 16, kt * 32, lane);
#pragma unroll
            for (int u = 0; u < 4; ++u)
                sacc[g * 4 + u] = wmma_bf16(a, bb[u], sacc[g * 4 + u]);
        }
        {
            v16bf bb = ld_frag(sK, 72, 12 * 16, kt * 32, lane);
            sacc[12] = wmma_bf16(a, bb, sacc[12]);
        }
    }
    const int ln = lane & 15;
    const int lh = (lane & 16) ? 8 : 0;
    float* Sw = sS + (size_t)w * 16 * 224;
#pragma unroll
    for (int j = 0; j < 13; ++j)
#pragma unroll
        for (int r = 0; r < 8; ++r)
            Sw[(lh + r) * 224 + j * 16 + ln] = sacc[j][r];
    __syncthreads();

    // ---- softmax: lanes 0..15 each own one row of this wave's 16-row tile
    if (lane < 16) {
        int r    = lane;
        int qrow = qt * 64 + w * 16 + r;
        bool valid = qrow < 196;
        float* Srow = Sw + r * 224;
        const float* brow = bias + ((size_t)h * 196 + (valid ? qrow : 0)) * 196;
        float mx = -1e30f;
        for (int j = 0; j < 196; ++j) {
            float s = Srow[j] * 0.125f + (valid ? brow[j] : 0.f);
            Srow[j] = s;
            mx = fmaxf(mx, s);
        }
        float sum = 0.f;
        for (int j = 0; j < 196; ++j) {
            float p = __expf(Srow[j] - mx);
            Srow[j] = p; sum += p;
        }
        float inv = 1.f / sum;
        unsigned short* Prow = (unsigned short*)Srow;  // bf16 pack in place
        for (int j = 0; j < 196; ++j) {
            float p = Srow[j] * inv;
            Prow[j] = f2bf(p);
        }
        for (int j = 196; j < 224; ++j) Prow[j] = 0;
    }
    __syncthreads();

    // ---- O = P * v (16x128 per wave, K=224)
    v8f oacc[8];
#pragma unroll
    for (int dj = 0; dj < 8; ++dj) oacc[dj] = v8f_zero();
    const unsigned short* Pb = (const unsigned short*)Sw;   // pitch 448 halves
#pragma unroll
    for (int kt = 0; kt < 7; ++kt) {
        v16bf a = ld_frag(Pb, 448, 0, kt * 32, lane);
#pragma unroll
        for (int g = 0; g < 2; ++g) {
            v16bf bb[4];
#pragma unroll
            for (int u = 0; u < 4; ++u)
                bb[u] = ld_frag(sV, 224, (g * 4 + u) * 16, kt * 32, lane);
#pragma unroll
            for (int u = 0; u < 4; ++u)
                oacc[g * 4 + u] = wmma_bf16(a, bb[u], oacc[g * 4 + u]);
        }
    }

    // ---- hardswish + store bf16 into hs[12544][2048]
#pragma unroll
    for (int dj = 0; dj < 8; ++dj)
#pragma unroll
        for (int r = 0; r < 8; ++r) {
            int qrow = qt * 64 + w * 16 + lh + r;
            if (qrow < 196) {
                float v  = oacc[dj][r];
                float hv = v * fminf(fmaxf(v + 3.f, 0.f), 6.f) * (1.f / 6.f);
                hs[((size_t)b * 196 + qrow) * 2048 + h * 128 + dj * 16 + ln] = f2bf(hv);
            }
        }
}

// ------------------------------------------------------------- final affine
__global__ void apply_bn_out(const float* __restrict__ y,
                             const float2* __restrict__ prm,
                             float* __restrict__ out, size_t n) {
    size_t i = (size_t)blockIdx.x * blockDim.x + threadIdx.x;
    size_t stride = (size_t)gridDim.x * blockDim.x;
    for (; i < n; i += stride) {
        float2 p = prm[i & 511];
        out[i] = p.x * y[i] + p.y;
    }
}

// ============================================================== launcher
extern "C" void kernel_launch(void* const* d_in, const int* in_sizes, int n_in,
                              void* d_out, int out_size, void* d_ws, size_t ws_size,
                              hipStream_t stream) {
    const float* x     = (const float*)d_in[0];
    const float* Wkv   = (const float*)d_in[1];
    const float* g_kv  = (const float*)d_in[2];
    const float* b_kv  = (const float*)d_in[3];
    const float* Wq    = (const float*)d_in[4];
    const float* g_q   = (const float*)d_in[5];
    const float* b_q   = (const float*)d_in[6];
    const float* bias  = (const float*)d_in[7];
    const float* Wproj = (const float*)d_in[8];
    const float* g_pr  = (const float*)d_in[9];
    const float* b_pr  = (const float*)d_in[10];
    float* out = (float*)d_out;

    char* ws = (char*)d_ws;
    size_t off = 0;
    auto alloc = [&](size_t bytes) -> char* {
        char* p = ws + off;
        off += (bytes + 255) & ~(size_t)255;
        return p;
    };
    unsigned short* xb    = (unsigned short*)alloc((size_t)12544 * 512 * 2);
    unsigned short* wkvb  = (unsigned short*)alloc((size_t)3072 * 512 * 2);
    unsigned short* wqb   = (unsigned short*)alloc((size_t)1024 * 512 * 2);
    unsigned short* wprb  = (unsigned short*)alloc((size_t)512 * 2048 * 2);
    float*  csum = (float*)alloc(4608 * 4);
    float*  csq  = (float*)alloc(4608 * 4);
    float2* prm  = (float2*)alloc(4608 * 8);
    unsigned short* ykv = (unsigned short*)alloc((size_t)12544 * 3072 * 2); // -> hs
    unsigned short* yq  = (unsigned short*)alloc((size_t)12544 * 1024 * 2); // -> yproj
    unsigned short* qbuf  = (unsigned short*)alloc((size_t)1024 * 256 * 64 * 2);
    unsigned short* kbuf  = (unsigned short*)alloc((size_t)1024 * 208 * 64 * 2);
    unsigned short* vtbuf = (unsigned short*)alloc((size_t)1024 * 128 * 224 * 2);
    unsigned short* hs    = ykv;           // reuse: ykv dead after repack
    float*          yproj = (float*)yq;    // reuse: yq dead after repack

    (void)in_sizes; (void)n_in; (void)out_size; (void)ws_size;

    const int ATTN_LDS = 9216 + 29952 + 57344 + 57344;   // 153,856 B
    hipFuncSetAttribute((const void*)attn_kernel,
                        hipFuncAttributeMaxDynamicSharedMemorySize, ATTN_LDS);

    zero_f32<<<(9216 + 255) / 256, 256, 0, stream>>>(csum, 9216);

    f32_to_bf16<<<2048, 256, 0, stream>>>(x,     xb,   (size_t)12544 * 512);
    f32_to_bf16<<<1024, 256, 0, stream>>>(Wkv,   wkvb, (size_t)3072 * 512);
    f32_to_bf16<<<512,  256, 0, stream>>>(Wq,    wqb,  (size_t)1024 * 512);
    f32_to_bf16<<<512,  256, 0, stream>>>(Wproj, wprb, (size_t)512 * 2048);

    gemm_bn<0><<<dim3(24, 98), 256, 0, stream>>>(xb, wkvb, ykv, nullptr,
                                                 csum, csq, 512, 3072);
    gemm_bn<0><<<dim3(8, 98), 256, 0, stream>>>(xb, wqb, yq, nullptr,
                                                csum + 3072, csq + 3072, 512, 1024);

    const float invM = 1.0f / 12544.0f;
    finalize_bn<<<(3072 + 255) / 256, 256, 0, stream>>>(csum, csq, g_kv, b_kv,
                                                        prm, 3072, invM);
    finalize_bn<<<(1024 + 255) / 256, 256, 0, stream>>>(csum + 3072, csq + 3072,
                                                        g_q, b_q, prm + 3072,
                                                        1024, invM);

    repack_qkv<<<4096, 256, 0, stream>>>(ykv, yq, prm, qbuf, kbuf, vtbuf);

    attn_kernel<<<dim3(4, 1024), 128, ATTN_LDS, stream>>>(qbuf, kbuf, vtbuf,
                                                          bias, hs);

    gemm_bn<1><<<dim3(4, 98), 256, 0, stream>>>(hs, wprb, nullptr, yproj,
                                                csum + 4096, csq + 4096, 2048, 512);

    finalize_bn<<<(512 + 255) / 256, 256, 0, stream>>>(csum + 4096, csq + 4096,
                                                       g_pr, b_pr, prm + 4096,
                                                       512, invM);

    apply_bn_out<<<4096, 256, 0, stream>>>(yproj, prm + 4096, out,
                                           (size_t)12544 * 512);
}
